// ContentGCN_5059471475267
// MI455X (gfx1250) — matrix operand order repaired and checked by reading, not compiled, not run on hardware
//
#include <hip/hip_runtime.h>
#include <hip/hip_bf16.h>

typedef _Float16 half_t;
typedef __attribute__((ext_vector_type(16))) _Float16 v16h;
typedef __attribute__((ext_vector_type(8)))  _Float16 v8h;
typedef __attribute__((ext_vector_type(8)))  float    v8f;

#define DIM   64
#define BC    128          // KV tile width
#define WAVES 8

// ---------------------------------------------------------------------------
// prep: per row of [B*H*N, 64] compute l2-normalized f16 copy (Q/K) and f16
// copy (V).  One thread per row; data is L2-resident so this pass is noise.
// ---------------------------------------------------------------------------
__global__ __launch_bounds__(256) void prep_kernel(const float* __restrict__ x,
                                                   half_t* __restrict__ q,
                                                   half_t* __restrict__ v,
                                                   int nrows)
{
    int row = blockIdx.x * 256 + threadIdx.x;
    if (row >= nrows) return;
    const float4* xr = (const float4*)(x + (size_t)row * DIM);
    float4 buf[16];
    float ss = 0.f;
#pragma unroll
    for (int i = 0; i < 16; ++i) {
        float4 b = xr[i];
        buf[i] = b;
        ss += b.x * b.x + b.y * b.y + b.z * b.z + b.w * b.w;
    }
    float inv = 1.0f / fmaxf(sqrtf(ss), 1e-12f);
    half_t* qr = q + (size_t)row * DIM;
    half_t* vr = v + (size_t)row * DIM;
#pragma unroll
    for (int i = 0; i < 16; ++i) {
        float4 b = buf[i];
        qr[4 * i + 0] = (half_t)(b.x * inv);
        qr[4 * i + 1] = (half_t)(b.y * inv);
        qr[4 * i + 2] = (half_t)(b.z * inv);
        qr[4 * i + 3] = (half_t)(b.w * inv);
        vr[4 * i + 0] = (half_t)b.x;
        vr[4 * i + 1] = (half_t)b.y;
        vr[4 * i + 2] = (half_t)b.z;
        vr[4 * i + 3] = (half_t)b.w;
    }
}

__global__ __launch_bounds__(256) void wcvt_kernel(const float* __restrict__ W1,
                                                   const float* __restrict__ W2,
                                                   half_t* __restrict__ w16a,
                                                   half_t* __restrict__ w16b)
{
    for (int i = threadIdx.x; i < DIM * DIM; i += 256) {
        w16a[i] = (half_t)W1[i];
        w16b[i] = (half_t)W2[i];
    }
}

// ---------------------------------------------------------------------------
// Flash-attention block:  out = relu( softmax(qn qn^T / a) x W^T + x )
// grid = (N/128, B*H), block = 256 threads (8 waves, 16 query rows per wave)
// KV tiles of 128; LDS = 16KB K + 16KB V^T + 32KB P staging = 64KB.
// ---------------------------------------------------------------------------
__global__ __launch_bounds__(256) void attn_kernel(
    const half_t* __restrict__ q16, const half_t* __restrict__ v16,
    const float* __restrict__ xin, const half_t* __restrict__ w16,
    const float* __restrict__ alpha_p, float* __restrict__ out, int N)
{
    __shared__ __align__(32) half_t Klds[BC * DIM];       // K tile, row-major [kv][d]
    __shared__ __align__(32) half_t VTlds[DIM * BC];      // V tile transposed [d][kv]
    __shared__ __align__(32) half_t Plds[WAVES * 16 * BC]; // per-wave C->A staging

    const int tid  = threadIdx.x;
    const int lane = tid & 31;
    const int wave = tid >> 5;
    const int hs   = lane >> 4;   // lane-half select (WMMA layout)
    const int l16  = lane & 15;
    const int rowBase = blockIdx.x * 128 + wave * 16;
    const size_t headOff = (size_t)blockIdx.y * (size_t)N;
    // softmax(S/alpha) == exp2((S - m) * log2(e)/alpha); fold scale into exponent
    const float k2 = 1.44269504088896f / fmaxf(alpha_p[0], 0.01f);
    half_t* pw = &Plds[wave * 16 * BC];

    // ---- Q fragments (A layout, 16 rows x 64 K, two 16x32 frags) ----
    v16h qa[2];
    {
        const half_t* qrow = q16 + (headOff + rowBase + l16) * DIM;
#pragma unroll
        for (int kf = 0; kf < 2; ++kf) {
            v8h lo = *(const v8h*)(qrow + kf * 32 + 16 * hs);
            v8h hi = *(const v8h*)(qrow + kf * 32 + 16 * hs + 8);
            v16h a;
#pragma unroll
            for (int i = 0; i < 8; ++i) { a[i] = lo[i]; a[8 + i] = hi[i]; }
            qa[kf] = a;
        }
    }

    const v8f zacc = {};
    float mrow[8], lrow[8];
    v8f O[4];
#pragma unroll
    for (int r = 0; r < 8; ++r) { mrow[r] = -1e30f; lrow[r] = 0.f; }
#pragma unroll
    for (int t = 0; t < 4; ++t) O[t] = zacc;

    for (int kv = 0; kv < N; kv += BC) {
        __syncthreads();
        // cooperative tile load: K row-major, V transposed (8192 halfs each)
#pragma unroll
        for (int i = 0; i < 4; ++i) {
            int idx8 = tid + i * 256;            // 1024 chunks of 8 halfs
            int row  = idx8 >> 3;                // 0..127
            int c8   = idx8 & 7;
            *(v8h*)&Klds[row * DIM + c8 * 8] =
                *(const v8h*)(q16 + (headOff + kv + row) * DIM + c8 * 8);
            v8h dv = *(const v8h*)(v16 + (headOff + kv + row) * DIM + c8 * 8);
#pragma unroll
            for (int j = 0; j < 8; ++j)
                VTlds[(c8 * 8 + j) * BC + row] = dv[j];
        }
        __syncthreads();

        // prefetch next KV tile into cache while we compute on this one
        {
            int kvn = kv + BC;
            if (kvn < N) {
                int r = tid & 127;
                const half_t* p = (tid < 128 ? q16 : v16) + (headOff + kvn + r) * DIM;
                __builtin_prefetch((const void*)p, 0, 3);
            }
        }

        // ---- S = Q K^T : 16 WMMAs (8 col tiles x 2 k-frags) ----
        v8f S[8];
#pragma unroll
        for (int t = 0; t < 8; ++t) {
            v8f acc = zacc;
#pragma unroll
            for (int kf = 0; kf < 2; ++kf) {
                v16h b = *(const v16h*)&Klds[(t * 16 + l16) * DIM + kf * 32 + 16 * hs];
                acc = __builtin_amdgcn_wmma_f32_16x16x32_f16(
                    false, qa[kf], false, b, (short)0, acc, false, false);
            }
            S[t] = acc;
        }

        // ---- online softmax (rows live in 16-lane halves) ----
        float scl[8];
#pragma unroll
        for (int r = 0; r < 8; ++r) {
            float v = fmaxf(fmaxf(fmaxf(S[0][r], S[1][r]), fmaxf(S[2][r], S[3][r])),
                            fmaxf(fmaxf(S[4][r], S[5][r]), fmaxf(S[6][r], S[7][r])));
            v = fmaxf(v, __shfl_xor(v, 1, 32));
            v = fmaxf(v, __shfl_xor(v, 2, 32));
            v = fmaxf(v, __shfl_xor(v, 4, 32));
            v = fmaxf(v, __shfl_xor(v, 8, 32));
            float mn = fmaxf(mrow[r], v);
            scl[r] = exp2f((mrow[r] - mn) * k2);
            mrow[r] = mn;
        }
#pragma unroll
        for (int t = 0; t < 8; ++t)
#pragma unroll
            for (int r = 0; r < 8; ++r) S[t][r] = exp2f((S[t][r] - mrow[r]) * k2);
#pragma unroll
        for (int r = 0; r < 8; ++r) {
            float s = ((S[0][r] + S[1][r]) + (S[2][r] + S[3][r])) +
                      ((S[4][r] + S[5][r]) + (S[6][r] + S[7][r]));
            s += __shfl_xor(s, 1, 32);
            s += __shfl_xor(s, 2, 32);
            s += __shfl_xor(s, 4, 32);
            s += __shfl_xor(s, 8, 32);
            lrow[r] = lrow[r] * scl[r] + s;
        }
#pragma unroll
        for (int t = 0; t < 4; ++t)
#pragma unroll
            for (int r = 0; r < 8; ++r) O[t][r] *= scl[r];

        // ---- P: C layout -> f16 row-major in per-wave LDS ----
#pragma unroll
        for (int r = 0; r < 8; ++r)
#pragma unroll
            for (int t = 0; t < 8; ++t)
                pw[(r + 8 * hs) * BC + t * 16 + l16] = (half_t)S[t][r];

        // re-read as A fragments (per-wave region, DS in-order)
        v16h pa[4];
#pragma unroll
        for (int kf = 0; kf < 4; ++kf) {
            v8h lo = *(const v8h*)&pw[l16 * BC + kf * 32 + 16 * hs];
            v8h hi = *(const v8h*)&pw[l16 * BC + kf * 32 + 16 * hs + 8];
            v16h a;
#pragma unroll
            for (int i = 0; i < 8; ++i) { a[i] = lo[i]; a[8 + i] = hi[i]; }
            pa[kf] = a;
        }

        // ---- O += P V : 16 WMMAs (4 d-tiles x 4 k-frags) ----
#pragma unroll
        for (int t = 0; t < 4; ++t) {
#pragma unroll
            for (int kf = 0; kf < 4; ++kf) {
                v16h b = *(const v16h*)&VTlds[(t * 16 + l16) * BC + kf * 32 + 16 * hs];
                O[t] = __builtin_amdgcn_wmma_f32_16x16x32_f16(
                    false, pa[kf], false, b, (short)0, O[t], false, false);
            }
        }
    }

    // ---- finalize softmax ----
#pragma unroll
    for (int r = 0; r < 8; ++r) {
        float inv_l = 1.0f / lrow[r];
#pragma unroll
        for (int t = 0; t < 4; ++t) O[t][r] *= inv_l;
    }

    // ---- linear: R = O W^T (8 WMMAs), via f16 staging of O ----
#pragma unroll
    for (int r = 0; r < 8; ++r)
#pragma unroll
        for (int t = 0; t < 4; ++t)
            pw[(r + 8 * hs) * DIM + t * 16 + l16] = (half_t)O[t][r];

    v16h oa[2];
#pragma unroll
    for (int kf = 0; kf < 2; ++kf) {
        v8h lo = *(const v8h*)&pw[l16 * DIM + kf * 32 + 16 * hs];
        v8h hi = *(const v8h*)&pw[l16 * DIM + kf * 32 + 16 * hs + 8];
        v16h a;
#pragma unroll
        for (int i = 0; i < 8; ++i) { a[i] = lo[i]; a[8 + i] = hi[i]; }
        oa[kf] = a;
    }

    v8f R[4];
#pragma unroll
    for (int t = 0; t < 4; ++t) {
        v8f acc = zacc;
#pragma unroll
        for (int kf = 0; kf < 2; ++kf) {
            // B[k=d][n=e] = W[e][d]; w16 row-major [e][d] -> contiguous along d
            v16h b = *(const v16h*)&w16[(t * 16 + l16) * DIM + kf * 32 + 16 * hs];
            acc = __builtin_amdgcn_wmma_f32_16x16x32_f16(
                false, oa[kf], false, b, (short)0, acc, false, false);
        }
        R[t] = acc;
    }

    // ---- residual + relu, fp32 ----
    const float* xrow = xin + (headOff + rowBase) * DIM;
    float* orow = out + (headOff + rowBase) * DIM;
#pragma unroll
    for (int r = 0; r < 8; ++r) {
        int m = r + 8 * hs;
#pragma unroll
        for (int t = 0; t < 4; ++t) {
            int c = t * 16 + l16;
            float val = R[t][r] + xrow[(size_t)m * DIM + c];
            orow[(size_t)m * DIM + c] = fmaxf(val, 0.f);
        }
    }
}

// ---------------------------------------------------------------------------
extern "C" void kernel_launch(void* const* d_in, const int* in_sizes, int n_in,
                              void* d_out, int out_size, void* d_ws, size_t ws_size,
                              hipStream_t stream)
{
    (void)in_sizes; (void)n_in; (void)out_size; (void)ws_size;
    const float* x  = (const float*)d_in[0];
    const float* W1 = (const float*)d_in[1];
    const float* W2 = (const float*)d_in[2];
    const float* a1 = (const float*)d_in[3];
    const float* a2 = (const float*)d_in[4];
    float* out = (float*)d_out;

    const int B = 4, H = 4, N = 4096;
    const int BHN = B * H * N;          // 65536 rows of 64

    char* ws = (char*)d_ws;
    half_t* q16  = (half_t*)ws;                                   // 8 MiB
    half_t* v16  = (half_t*)(ws + (size_t)8  * 1024 * 1024);      // 8 MiB
    float*  x1   = (float*) (ws + (size_t)16 * 1024 * 1024);      // 16 MiB
    half_t* w16a = (half_t*)(ws + (size_t)32 * 1024 * 1024);      // 8 KiB
    half_t* w16b = w16a + DIM * DIM;                              // 8 KiB

    dim3 agrid(N / 128, B * H);

    // block 1
    prep_kernel<<<BHN / 256, 256, 0, stream>>>(x, q16, v16, BHN);
    wcvt_kernel<<<1, 256, 0, stream>>>(W1, W2, w16a, w16b);
    attn_kernel<<<agrid, 256, 0, stream>>>(q16, v16, x, w16a, a1, x1, N);
    // block 2
    prep_kernel<<<BHN / 256, 256, 0, stream>>>(x1, q16, v16, BHN);
    attn_kernel<<<agrid, 256, 0, stream>>>(q16, v16, x1, w16b, a2, out, N);
}
